// CustomBidirectionalRNN_3075196584537
// MI455X (gfx1250) — matrix operand order repaired and checked by reading, not compiled, not run on hardware
//
#include <hip/hip_runtime.h>
#include <hip/hip_bf16.h>

// Bidirectional LSTM, B=64 T=1024 D=512 H=768.
// Per time step: z = [x_t | h_prev](64x1280, bf16) @ WcatT^T(1280x3072, bf16)
// via v_wmma_f32_16x16x32_bf16 (fp32 accum), fused gates + state update.
// Both directions per launch (gridDim.z=2); 1024 step launches.

#define Bb   64
#define Tt   1024
#define Dd   512
#define Hh   768
#define K4   3072            // 4*H
#define KK   1280            // D+H
#define KC   256             // K chunk staged in LDS
#define ASTR (KC + 8)        // LDS row stride (bf16 elems), rows 16B-aligned
#define ZSTR 132             // LDS z row stride (floats)

typedef __attribute__((ext_vector_type(16))) __bf16        v16bf;
typedef __attribute__((ext_vector_type(8)))  float         v8f;
typedef __attribute__((ext_vector_type(4)))  unsigned int  v4u;

union FragU { v4u q[2]; v16bf v; };

__device__ __forceinline__ unsigned short f2bf(float f) {
  unsigned int u = __builtin_bit_cast(unsigned int, f);
  unsigned int r = u + 0x7FFFu + ((u >> 16) & 1u);   // RNE
  return (unsigned short)(r >> 16);
}

// ---------------- prep kernels ----------------

// xbf[t][b][d] = bf16( x[b][t][d] )   (time-major rows for contiguous LDS staging)
__global__ void cvt_x_kernel(const float* __restrict__ x, unsigned short* __restrict__ xbf) {
  size_t i = (size_t)blockIdx.x * 256 + threadIdx.x;
  if (i >= (size_t)Tt * Bb * Dd) return;
  int d = (int)(i % Dd);
  size_t tb = i / Dd;
  int b = (int)(tb % Bb);
  int t = (int)(tb / Bb);
  xbf[i] = f2bf(x[((size_t)b * Tt + t) * Dd + d]);
}

// WcatT[dir][n][k] = bf16( k<D ? W[k][n] : U[k-D][n] )  — B stored column-major-of-B
__global__ void build_w_kernel(const float* __restrict__ Wf, const float* __restrict__ Uf,
                               const float* __restrict__ Wb, const float* __restrict__ Ub,
                               unsigned short* __restrict__ wcatT) {
  size_t i = (size_t)blockIdx.x * 256 + threadIdx.x;
  if (i >= (size_t)2 * K4 * KK) return;
  int k = (int)(i % KK);
  size_t dn = i / KK;
  int n = (int)(dn % K4);
  int dir = (int)(dn / K4);
  const float* W = dir ? Wb : Wf;
  const float* U = dir ? Ub : Uf;
  float v = (k < Dd) ? W[(size_t)k * K4 + n] : U[(size_t)(k - Dd) * K4 + n];
  wcatT[i] = f2bf(v);
}

__global__ void init_state_kernel(unsigned short* __restrict__ hbf, float* __restrict__ cst) {
  int i = blockIdx.x * 256 + threadIdx.x;
  if (i < 2 * 2 * Bb * Hh) hbf[i] = 0;     // bf16 +0.0
  if (i < 2 * Bb * Hh)     cst[i] = 0.0f;
}

// ---------------- fused LSTM step ----------------
// grid (24,1,2): x = 32-unit group, z = direction. block = 256 (8 waves).
// Wave w: gate = w>>1, sub = w&1 -> N tile = gate*768 + u0 + sub*16, all 4 M tiles.
__global__ __launch_bounds__(256) void lstm_step_kernel(
    const unsigned short* __restrict__ xbf,    // [T][B][D] bf16
    const unsigned short* __restrict__ wcatT,  // [2][3072][1280] bf16
    const float* __restrict__ bias_f,          // [3072]
    const float* __restrict__ bias_b,          // [3072]
    unsigned short* __restrict__ hbf,          // [2 pp][2 dir][B][H] bf16
    float* __restrict__ cst,                   // [2 dir][B][H]
    float* __restrict__ out,                   // [B][T][2H]
    int t)
{
  extern __shared__ char smem_raw[];
  unsigned short* Alds = (unsigned short*)smem_raw;   // 64 x ASTR bf16 (33792 B)
  float*          Zlds = (float*)smem_raw;            // 64 x ZSTR f32  (33792 B), reused

  const int dir  = blockIdx.z;
  const int u0   = blockIdx.x * 32;
  const int tid  = threadIdx.x;
  const int lane = tid & 31;
  const int wave = tid >> 5;
  const int half = lane >> 4;       // which 16-lane half
  const int l16  = lane & 15;
  const int pp   = t & 1;

  const int teff = (dir == 0) ? t : (Tt - 1 - t);

  const unsigned short* __restrict__ xsrc  = xbf + (size_t)teff * Bb * Dd;
  const unsigned short* __restrict__ hprev = hbf + (size_t)(pp * 2 + dir) * (Bb * Hh);
  unsigned short* __restrict__ hnext       = hbf + (size_t)(((pp ^ 1) * 2) + dir) * (Bb * Hh);

  const int gate  = wave >> 1;
  const int sub   = wave & 1;
  const int ntile = gate * Hh + u0 + sub * 16;
  // B fragment base: lane's column of WcatT, contiguous in K
  const unsigned short* __restrict__ bbase =
      wcatT + ((size_t)dir * K4 + ntile + l16) * KK + half * 16;

  v8f acc[4];
  #pragma unroll
  for (int m = 0; m < 4; ++m)
    #pragma unroll
    for (int r = 0; r < 8; ++r) acc[m][r] = 0.0f;

  for (int kc = 0; kc < KK / KC; ++kc) {
    const int kbase = kc * KC;
    __syncthreads();
    // stage A chunk [64 x 256] bf16 into LDS (each chunk is homogeneous: x or h)
    for (int v = tid; v < 64 * (KC / 8); v += 256) {
      int row = v >> 5;
      int cv  = v & 31;
      int col = kbase + cv * 8;
      const unsigned short* src = (col < Dd) ? (xsrc + (size_t)row * Dd + col)
                                             : (hprev + (size_t)row * Hh + (col - Dd));
      *(v4u*)&Alds[row * ASTR + cv * 8] = *(const v4u*)src;
    }
    __syncthreads();

    #pragma unroll
    for (int kk = 0; kk < KC / 32; ++kk) {
      // B fragment: per lane k = half*16 + i (i=0..15): contiguous 32 bytes
      FragU bfr;
      const unsigned short* bp = bbase + (kbase + kk * 32);
      bfr.q[0] = *(const v4u*)(bp);
      bfr.q[1] = *(const v4u*)(bp + 8);
      // All 4 A fragments into DISTINCT registers first (A layout matching B:
      // per lane k runs half*16..+15 contiguous), so the 4 WMMAs can
      // issue back-to-back after a single dscnt wait.
      FragU afr[4];
      #pragma unroll
      for (int m = 0; m < 4; ++m) {
        const unsigned short* ap = &Alds[(m * 16 + l16) * ASTR + kk * 32 + half * 16];
        afr[m].q[0] = *(const v4u*)(ap);
        afr[m].q[1] = *(const v4u*)(ap + 8);
      }
      #pragma unroll
      for (int m = 0; m < 4; ++m)
        acc[m] = __builtin_amdgcn_wmma_f32_16x16x32_bf16(
            false, afr[m].v, false, bfr.v, (short)0, acc[m], false, false);
    }
  }

  __syncthreads();   // all waves done with Alds -> reuse as Zlds
  // C/D layout: VGPR r -> row m*16 + half*8 + r, col = l16; local z col = gate*32 + ul
  #pragma unroll
  for (int m = 0; m < 4; ++m)
    #pragma unroll
    for (int r = 0; r < 8; ++r)
      Zlds[(m * 16 + half * 8 + r) * ZSTR + wave * 16 + l16] = acc[m][r];
  __syncthreads();

  // fused gates + state update: thread owns unit ul = tid&31, rows wave + 8j
  const float* __restrict__ bias = (dir == 0) ? bias_f : bias_b;
  const int ul = tid & 31;
  const int u  = u0 + ul;
  const float bi  = bias[u];
  const float bf_ = bias[Hh + u];
  const float bg  = bias[2 * Hh + u];
  const float bo  = bias[3 * Hh + u];
  #pragma unroll
  for (int j = 0; j < 8; ++j) {
    int row = (tid >> 5) + j * 8;
    float zi = Zlds[row * ZSTR + 0 * 32 + ul] + bi;
    float zf = Zlds[row * ZSTR + 1 * 32 + ul] + bf_;
    float zg = Zlds[row * ZSTR + 2 * 32 + ul] + bg;
    float zo = Zlds[row * ZSTR + 3 * 32 + ul] + bo;
    float ig = 1.0f / (1.0f + __expf(-zi));
    float fg = 1.0f / (1.0f + __expf(-zf));
    float gg = tanhf(zg);
    float og = 1.0f / (1.0f + __expf(-zo));
    size_t cidx = ((size_t)dir * Bb + row) * Hh + u;
    float cc = fg * cst[cidx] + ig * gg;
    cst[cidx] = cc;
    float hh = og * tanhf(cc);
    out[((size_t)row * Tt + teff) * (2 * Hh) + (size_t)dir * Hh + u] = hh;
    hnext[(size_t)row * Hh + u] = f2bf(hh);
  }
}

// ---------------- launcher ----------------
extern "C" void kernel_launch(void* const* d_in, const int* in_sizes, int n_in,
                              void* d_out, int out_size, void* d_ws, size_t ws_size,
                              hipStream_t stream) {
  const float* x  = (const float*)d_in[0];
  const float* Wf = (const float*)d_in[1];
  const float* Uf = (const float*)d_in[2];
  const float* bf = (const float*)d_in[3];
  const float* Wb = (const float*)d_in[4];
  const float* Ub = (const float*)d_in[5];
  const float* bb = (const float*)d_in[6];
  float* out = (float*)d_out;

  char* ws = (char*)d_ws;
  const size_t SZ_WCAT = (size_t)2 * K4 * KK * 2;        // 15,728,640
  const size_t SZ_X    = (size_t)Tt * Bb * Dd * 2;       // 67,108,864
  const size_t SZ_H    = (size_t)2 * 2 * Bb * Hh * 2;    //    393,216
  unsigned short* wcatT = (unsigned short*)(ws);
  unsigned short* xbf   = (unsigned short*)(ws + SZ_WCAT);
  unsigned short* hbf   = (unsigned short*)(ws + SZ_WCAT + SZ_X);
  float*          cst   = (float*)(ws + SZ_WCAT + SZ_X + SZ_H);

  {
    size_t n = (size_t)Tt * Bb * Dd;
    cvt_x_kernel<<<(unsigned)((n + 255) / 256), 256, 0, stream>>>(x, xbf);
  }
  {
    size_t n = (size_t)2 * K4 * KK;
    build_w_kernel<<<(unsigned)((n + 255) / 256), 256, 0, stream>>>(Wf, Uf, Wb, Ub, wcatT);
  }
  init_state_kernel<<<(2 * 2 * Bb * Hh + 255) / 256, 256, 0, stream>>>(hbf, cst);

  const size_t ldsBytes = 64 * ASTR * sizeof(unsigned short);  // == 64*ZSTR*4 == 33792
  for (int t = 0; t < Tt; ++t) {
    lstm_step_kernel<<<dim3(24, 1, 2), 256, ldsBytes, stream>>>(
        xbf, wcatT, bf, bb, hbf, cst, out, t);
  }
}
